// MultiHeadAttention_33749853012523
// MI455X (gfx1250) — matrix operand order repaired and checked by reading, not compiled, not run on hardware
//
#include <hip/hip_runtime.h>
#include <hip/hip_bf16.h>

typedef __bf16 bf16_t;
typedef __attribute__((ext_vector_type(16))) __bf16 v16bf;
typedef __attribute__((ext_vector_type(8)))  float  v8f;
typedef __attribute__((ext_vector_type(4)))  int    v4i;

namespace {
constexpr int Bsz   = 4;
constexpr int Lseq  = 1024;
constexpr int Dm    = 1024;
constexpr int Hh    = 16;
constexpr int HD    = 64;
constexpr int BH    = Bsz * Hh;     // 64
constexpr int Mrows = Bsz * Lseq;   // 4096
constexpr int PE_ROWS = 1056;       // rel indices used: [1,1024]; pad to 1056
constexpr float INV_SCALE = 0.125f; // 1/sqrt(64)
}

struct FragAB { union { v16bf v; unsigned u[8]; unsigned short h[16]; bf16_t b[16]; }; };
struct FragC  { union { v8f v; float f[8]; }; };

// 16-bit A-matrix (16x32) K pattern per lane-half (ISA 7.12.2) — used for LDS prob gathers
__device__ __forceinline__ int kpatA(int v, int half) {
  int base = (v < 4) ? (2 * v) : (16 + 2 * (v - 4));
  return base + 8 * half;
}

__device__ __forceinline__ v8f wmma_bf16(v16bf a, v16bf b, v8f c) {
  return __builtin_amdgcn_wmma_f32_16x16x32_bf16(false, a, false, b, (short)0, c, false, false);
}

// ---- CDNA5 async global->LDS copy (b128), with portable fallback ----
#if defined(__AMDGCN__) && __has_builtin(__builtin_amdgcn_global_load_async_to_lds_b128)
#define HAVE_ASYNC 1
#else
#define HAVE_ASYNC 0
#endif

typedef __attribute__((address_space(1))) v4i gv4i;
typedef __attribute__((address_space(3))) v4i lv4i;

__device__ __forceinline__ void async_cp16(const void* g, void* l) {
#if HAVE_ASYNC
  __builtin_amdgcn_global_load_async_to_lds_b128((gv4i*)g, (lv4i*)l, 0, 0);
#else
  *(uint4*)l = *(const uint4*)g;
#endif
}
__device__ __forceinline__ void async_wait_all() {
#if HAVE_ASYNC
#if __has_builtin(__builtin_amdgcn_s_wait_asynccnt)
  __builtin_amdgcn_s_wait_asynccnt(0);
#else
  asm volatile("s_wait_asynccnt 0x0" ::: "memory");
#endif
#endif
}

// ---------------- conversion kernels (all layouts chosen so WMMA B-operands are
// column-major-in-N, i.e. fragment loads are contiguous b128s) ----------------
__global__ void cvt_f32_bf16(const float* __restrict__ in, bf16_t* __restrict__ out, int n) {
  int i = blockIdx.x * blockDim.x + threadIdx.x;
  if (i < n) out[i] = (bf16_t)in[i];
}

// pevT[d][j] = pev[j][d]
__global__ void cvt_pev_T(const float* __restrict__ pev, bf16_t* __restrict__ out) {
  int i = blockIdx.x * blockDim.x + threadIdx.x;
  if (i < HD * PE_ROWS) {
    int d = i / PE_ROWS, j = i % PE_ROWS;
    out[i] = (bf16_t)pev[j * HD + d];
  }
}

// ---------------- tiled bf16 WMMA GEMM: C[4096][1024] = A @ W^T ----------------
// A: bf16 row-major [M][K].  Bh: bf16 = W natural [n][k] (column-major B operand).
// MODE 0: out bf16 [bh][s][d]   (Q, K head split)
// MODE 1: out bf16 [bh][d][s]   (V transposed per head)
// MODE 3: out f32 row-major [M][N]
template <int MODE>
__global__ __launch_bounds__(256) void gemm_bf16_wmma(
    const bf16_t* __restrict__ A, const bf16_t* __restrict__ Bh, void* __restrict__ Cout)
{
  constexpr int K = Dm, N = Dm, KT = 64;
  __shared__ alignas(16) bf16_t AsBuf[2][128 * KT];   // [m][k]
  __shared__ alignas(16) bf16_t BsBuf[2][128 * KT];   // [n][k]
  const int tid  = threadIdx.x;
  const int lane = tid & 31, w = tid >> 5;
  const int nL   = lane & 15, half = lane >> 4;
  const int bm = blockIdx.x * 128, bn = blockIdx.y * 128;
  const int wm = (w >> 1) * 32, wn = (w & 1) * 64;

  FragC acc[2][4];
#pragma unroll
  for (int mt = 0; mt < 2; ++mt)
#pragma unroll
    for (int nt = 0; nt < 4; ++nt)
#pragma unroll
      for (int i = 0; i < 8; ++i) acc[mt][nt].f[i] = 0.0f;

  auto stage = [&](int buf, int k0) {   // 16 KB per matrix, 4 b128 chunks/thread each
#pragma unroll
    for (int r = 0; r < 4; ++r) {
      int c = tid + r * 256;            // uint4 chunk id (1024 per matrix)
      int row = c >> 3, cc = (c & 7) * 8;
      async_cp16(&A[(size_t)(bm + row) * K + k0 + cc], &AsBuf[buf][c * 8]);
      async_cp16(&Bh[(size_t)(bn + row) * K + k0 + cc], &BsBuf[buf][c * 8]);
    }
  };

  stage(0, 0);                          // prologue of double-buffered pipeline
  for (int k0 = 0; k0 < K; k0 += KT) {
    const int buf = (k0 >> 6) & 1;
    async_wait_all();
    __syncthreads();
    if (k0 + KT < K) stage(buf ^ 1, k0 + KT);   // prefetch next tile while computing
    const bf16_t* Asb = AsBuf[buf];
    const bf16_t* Bsb = BsBuf[buf];
#pragma unroll
    for (int ks = 0; ks < KT; ks += 32) {
      const int t0 = ks >> 3;
      FragAB a[2], b[4];
#pragma unroll
      for (int mt = 0; mt < 2; ++mt) {  // A frag: two contiguous b128 runs
        const uint4* rp = (const uint4*)&Asb[(wm + mt * 16 + nL) * KT];
        *(uint4*)&a[mt].u[0] = rp[t0 + half];
        *(uint4*)&a[mt].u[4] = rp[t0 + 2 + half];
      }
#pragma unroll
      for (int nt = 0; nt < 4; ++nt) {  // B frag: lane = column, K contiguous
        const uint4* rp = (const uint4*)&Bsb[(wn + nt * 16 + nL) * KT];
        *(uint4*)&b[nt].u[0] = rp[t0 + half];
        *(uint4*)&b[nt].u[4] = rp[t0 + 2 + half];
      }
#pragma unroll
      for (int mt = 0; mt < 2; ++mt)
#pragma unroll
        for (int nt = 0; nt < 4; ++nt)
          acc[mt][nt].v = wmma_bf16(a[mt].v, b[nt].v, acc[mt][nt].v);
    }
  }

#pragma unroll
  for (int mt = 0; mt < 2; ++mt)
#pragma unroll
    for (int nt = 0; nt < 4; ++nt)
#pragma unroll
      for (int v = 0; v < 8; ++v) {
        int m   = v + 8 * half;
        int row = bm + wm + mt * 16 + m;
        int col = bn + wn + nt * 16 + nL;
        float val = acc[mt][nt].f[v];
        if constexpr (MODE == 3) {
          ((float*)Cout)[(size_t)row * N + col] = val;
        } else {
          int bb = row >> 10, s = row & 1023;
          int hh = col >> 6,  d = col & 63;
          bf16_t* O = (bf16_t*)Cout;
          if constexpr (MODE == 1)
            O[((size_t)(bb * Hh + hh) * HD + d) * Lseq + s] = (bf16_t)val;   // V^T
          else
            O[((size_t)(bb * Hh + hh) * Lseq + s) * HD + d] = (bf16_t)val;   // Q / K
        }
      }
}

// ---------------- attention with relative position bias ----------------
// One wave per 16 query rows of one (b,h). LDS: S[16][1024] probs, T[16][1056] rel scores.
__global__ __launch_bounds__(32) void attn_rel_kernel(
    const bf16_t* __restrict__ Qh, const bf16_t* __restrict__ Kh,
    const bf16_t* __restrict__ Vt, const bf16_t* __restrict__ pekh,
    const bf16_t* __restrict__ pevT, bf16_t* __restrict__ outh)
{
  extern __shared__ float smem[];
  float* S  = smem;               // [16][Lseq]  scores -> probabilities
  float* Tb = smem + 16 * Lseq;   // [16][TW]    T[q][j] = Q . pek[j]
  constexpr int TW = PE_ROWS;

  const int blk = blockIdx.x;
  const int qt  = blk & 63;
  const int bh  = blk >> 6;
  const int qb  = qt * 16;
  const int lane = threadIdx.x & 31;
  const int nL = lane & 15, half = lane >> 4;

  const bf16_t* Qp  = Qh + (size_t)bh * Lseq * HD;   // [s][d]
  const bf16_t* Kp  = Kh + (size_t)bh * Lseq * HD;   // [k][d]
  const bf16_t* Vtp = Vt + (size_t)bh * HD * Lseq;   // [d][k]

  // Q tile 16x64 as two A fragments; each = 2 contiguous b128 loads
  FragAB qf[2];
  {
    const uint4* qrow = (const uint4*)&Qp[(size_t)(qb + nL) * HD];
#pragma unroll
    for (int kk = 0; kk < 2; ++kk) {
      *(uint4*)&qf[kk].u[0] = qrow[4 * kk + half];
      *(uint4*)&qf[kk].u[4] = qrow[4 * kk + 2 + half];
    }
  }

  int jlo = Lseq - (qb + 15); if (jlo < 0) jlo = 0;
  const int jbase = jlo & ~15;

  // One 16-col tile of (Q @ Brows^T): Brows is [col][64] row-major (column-major B operand)
  auto mm_tile = [&](const bf16_t* Brows, int col0) -> FragC {
    FragC a;
#pragma unroll
    for (int i = 0; i < 8; ++i) a.f[i] = 0.0f;
    const uint4* brow = (const uint4*)&Brows[(size_t)(col0 + nL) * HD];
#pragma unroll
    for (int kk = 0; kk < 2; ++kk) {
      FragAB bfr;
      *(uint4*)&bfr.u[0] = brow[4 * kk + half];
      *(uint4*)&bfr.u[4] = brow[4 * kk + 2 + half];
      a.v = wmma_bf16(qf[kk].v, bfr.v, a.v);
    }
    return a;
  };
  auto store_T = [&](const FragC& a, int jb) {
#pragma unroll
    for (int v = 0; v < 8; ++v) {
      int m = v + 8 * half;
      Tb[m * TW + (jb - jbase) + nL] = a.f[v];
    }
  };

  // Phase 1: T = Q @ pek^T — two independent tiles in flight to fill WMMA hazard slots
  {
    int jb = jbase;
    for (; jb + 16 <= Lseq; jb += 32) {
      FragC a0 = mm_tile(pekh, jb);
      FragC a1 = mm_tile(pekh, jb + 16);
      store_T(a0, jb);
      store_T(a1, jb + 16);
    }
    for (; jb <= Lseq; jb += 16) {
      FragC a0 = mm_tile(pekh, jb);
      store_T(a0, jb);
    }
  }

  const int kend = qb + 16;

  auto store_scores = [&](const FragC& a, int kb) {
#pragma unroll
    for (int v = 0; v < 8; ++v) {
      int m = v + 8 * half;
      int q = qb + m, k = kb + nL;
      float s = -__builtin_inff();
      if (k <= q) s = (a.f[v] + Tb[m * TW + (k - q + Lseq - jbase)]) * INV_SCALE;
      S[m * Lseq + k] = s;
    }
  };

  // Phase 2: S = (Q K^T + T[skewed]) / SCALE with causal mask — paired tiles
  {
    int kb = 0;
    for (; kb + 32 <= kend; kb += 32) {
      FragC a0 = mm_tile(Kp, kb);
      FragC a1 = mm_tile(Kp, kb + 16);
      store_scores(a0, kb);
      store_scores(a1, kb + 16);
    }
    for (; kb < kend; kb += 16) {
      FragC a0 = mm_tile(Kp, kb);
      store_scores(a0, kb);
    }
  }

  // Phase 3: row softmax (wave32 shuffles)
  for (int m = 0; m < 16; ++m) {
    float* row = S + m * Lseq;
    float mx = -__builtin_inff();
    for (int k = lane; k < kend; k += 32) mx = fmaxf(mx, row[k]);
#pragma unroll
    for (int off = 16; off; off >>= 1) mx = fmaxf(mx, __shfl_xor(mx, off, 32));
    float sum = 0.0f;
    for (int k = lane; k < kend; k += 32) {
      float p = __expf(row[k] - mx);
      row[k] = p; sum += p;
    }
#pragma unroll
    for (int off = 16; off; off >>= 1) sum += __shfl_xor(sum, off, 32);
    float inv = 1.0f / sum;
    for (int k = lane; k < kend; k += 32) row[k] *= inv;
  }

  FragC o[4];
#pragma unroll
  for (int nt = 0; nt < 4; ++nt)
#pragma unroll
    for (int i = 0; i < 8; ++i) o[nt].f[i] = 0.0f;

  // Phase 4a: O += P @ V  (V^T [d][k] is column-major B => b128 frags; 4 independent accs)
  for (int kb = 0; kb < kend; kb += 32) {
    FragAB af;
#pragma unroll
    for (int v = 0; v < 8; ++v) {
      int kk = kb + kpatA(v, half);
      float p0 = (kk < kend)     ? S[nL * Lseq + kk]     : 0.0f;
      float p1 = (kk + 1 < kend) ? S[nL * Lseq + kk + 1] : 0.0f;
      af.b[2 * v]     = (bf16_t)p0;
      af.b[2 * v + 1] = (bf16_t)p1;
    }
#pragma unroll
    for (int nt = 0; nt < 4; ++nt) {
      FragAB bfr;
      const bf16_t* vrow = &Vtp[(size_t)(nt * 16 + nL) * Lseq];
      *(uint4*)&bfr.u[0] = *(const uint4*)&vrow[kb + 8 * half];
      *(uint4*)&bfr.u[4] = *(const uint4*)&vrow[kb + 16 + 8 * half];
      o[nt].v = wmma_bf16(af.v, bfr.v, o[nt].v);
    }
  }

  // Phase 4b: O += P2 @ pev, P2[q][j] = P[q][j + q - 1024] (skewed LDS gather; pev^T [d][j])
  for (int jb = jbase; jb < 1040; jb += 32) {
    FragAB af;
    int q = qb + nL;
#pragma unroll
    for (int v = 0; v < 8; ++v) {
      int j0 = jb + kpatA(v, half);
      int k0 = j0 + q - Lseq;
      float p0 = (j0 <= Lseq && k0 >= 0)         ? S[nL * Lseq + k0]     : 0.0f;
      float p1 = (j0 + 1 <= Lseq && k0 + 1 >= 0) ? S[nL * Lseq + k0 + 1] : 0.0f;
      af.b[2 * v]     = (bf16_t)p0;
      af.b[2 * v + 1] = (bf16_t)p1;
    }
#pragma unroll
    for (int nt = 0; nt < 4; ++nt) {
      FragAB bfr;
      const bf16_t* pr = &pevT[(size_t)(nt * 16 + nL) * PE_ROWS];
      *(uint4*)&bfr.u[0] = *(const uint4*)&pr[jb + 8 * half];
      *(uint4*)&bfr.u[4] = *(const uint4*)&pr[jb + 16 + 8 * half];
      o[nt].v = wmma_bf16(af.v, bfr.v, o[nt].v);
    }
  }

  // Store: outh[b*L + q][h*64 + d] bf16 (row-major A for the Wo GEMM)
  const int bb = bh >> 4, hh = bh & 15;
#pragma unroll
  for (int nt = 0; nt < 4; ++nt)
#pragma unroll
    for (int v = 0; v < 8; ++v) {
      int m = v + 8 * half;
      int d = nt * 16 + nL;
      outh[(size_t)(bb * Lseq + qb + m) * Dm + hh * HD + d] = (bf16_t)o[nt].f[v];
    }
}

// ---------------- host ----------------
extern "C" void kernel_launch(void* const* d_in, const int* in_sizes, int n_in,
                              void* d_out, int out_size, void* d_ws, size_t ws_size,
                              hipStream_t stream) {
  (void)in_sizes; (void)n_in; (void)out_size; (void)ws_size;
  const float* x   = (const float*)d_in[0];
  const float* Wq  = (const float*)d_in[1];
  const float* Wk  = (const float*)d_in[2];
  const float* Wv  = (const float*)d_in[3];
  const float* Wo  = (const float*)d_in[4];
  const float* pek = (const float*)d_in[5];
  const float* pev = (const float*)d_in[6];

  char* ws = (char*)d_ws;
  size_t off = 0;
  auto alloc = [&](size_t bytes) -> void* {
    void* p = ws + off;
    off = (off + bytes + 255) & ~(size_t)255;
    return p;
  };
  bf16_t* xh   = (bf16_t*)alloc((size_t)Mrows * Dm * 2);
  bf16_t* wqh  = (bf16_t*)alloc((size_t)Dm * Dm * 2);   // W natural [n][k] in bf16
  bf16_t* wkh  = (bf16_t*)alloc((size_t)Dm * Dm * 2);
  bf16_t* wvh  = (bf16_t*)alloc((size_t)Dm * Dm * 2);
  bf16_t* woh  = (bf16_t*)alloc((size_t)Dm * Dm * 2);
  bf16_t* pekh = (bf16_t*)alloc((size_t)PE_ROWS * HD * 2);  // natural [j][d]
  bf16_t* pevT = (bf16_t*)alloc((size_t)HD * PE_ROWS * 2);  // transposed [d][j]
  bf16_t* Qh   = (bf16_t*)alloc((size_t)BH * Lseq * HD * 2);
  bf16_t* Kh   = (bf16_t*)alloc((size_t)BH * Lseq * HD * 2);
  bf16_t* Vt   = (bf16_t*)alloc((size_t)BH * HD * Lseq * 2);
  bf16_t* outh = (bf16_t*)alloc((size_t)Mrows * Dm * 2);

  const int nx = Mrows * Dm, nw = Dm * Dm, np = PE_ROWS * HD;
  cvt_f32_bf16<<<(nx + 255) / 256, 256, 0, stream>>>(x,  xh,  nx);
  cvt_f32_bf16<<<(nw + 255) / 256, 256, 0, stream>>>(Wq, wqh, nw);
  cvt_f32_bf16<<<(nw + 255) / 256, 256, 0, stream>>>(Wk, wkh, nw);
  cvt_f32_bf16<<<(nw + 255) / 256, 256, 0, stream>>>(Wv, wvh, nw);
  cvt_f32_bf16<<<(nw + 255) / 256, 256, 0, stream>>>(Wo, woh, nw);
  cvt_f32_bf16<<<(np + 255) / 256, 256, 0, stream>>>(pek, pekh, np);
  cvt_pev_T  <<<(np + 255) / 256, 256, 0, stream>>>(pev, pevT);

  dim3 gg(Mrows / 128, Dm / 128);
  gemm_bf16_wmma<0><<<gg, 256, 0, stream>>>(xh, wqh, Qh);
  gemm_bf16_wmma<0><<<gg, 256, 0, stream>>>(xh, wkh, Kh);
  gemm_bf16_wmma<1><<<gg, 256, 0, stream>>>(xh, wvh, Vt);

  size_t smem = (size_t)(16 * Lseq + 16 * PE_ROWS) * sizeof(float); // 130 KB (<320 KB/WGP)
  (void)hipFuncSetAttribute((const void*)attn_rel_kernel,
                            hipFuncAttributeMaxDynamicSharedMemorySize, (int)smem);
  attn_rel_kernel<<<BH * (Lseq / 16), 32, smem, stream>>>(Qh, Kh, Vt, pekh, pevT, outh);

  gemm_bf16_wmma<3><<<gg, 256, 0, stream>>>(outh, woh, d_out);
}
